// LiquidEmbedding_60885456388528
// MI455X (gfx1250) — matrix operand order repaired and checked by reading, not compiled, not run on hardware
//
#include <hip/hip_runtime.h>

// ---------------------------------------------------------------------------
// LiquidEmbedding pipeline for gfx1250 (MI455X), wave32, WMMA bf16 +
// async global->LDS staging (ASYNCcnt) with double buffering.
// Stages: embed -> conv1 (GEMM+ReLU) -> conv2 (GEMM+ReLU) -> FFT(2048) ->
//         spectral threshold -> masked iFFT -> projection GEMM. loss == 0.
// ---------------------------------------------------------------------------

typedef __attribute__((ext_vector_type(16))) __bf16 v16bf;
typedef __attribute__((ext_vector_type(8)))  float  v8f;

#define Bn 8
#define Ln 2048
#define En 512
#define Kc 1536          // 3*En, conv GEMM depth
#define LP (Ln + 2)      // padded rows per batch (zero row top+bottom)

union FragBF { v16bf v; uint4 q[2]; };

static __device__ __forceinline__ unsigned short f2bf(float f) {
    unsigned int u = __float_as_uint(f);
    u += 0x7fffu + ((u >> 16) & 1u);          // round-to-nearest-even
    return (unsigned short)(u >> 16);
}

// Async copy of 16 bytes global -> LDS (GLOBAL_LOAD_ASYNC_TO_LDS_B128).
static __device__ __forceinline__ void async_copy16(unsigned lds_off,
                                                    const void* gaddr) {
    asm volatile("global_load_async_to_lds_b128 %0, %1, off"
                 :: "v"(lds_off), "v"(gaddr) : "memory");
}
static __device__ __forceinline__ void wait_async0() {
    asm volatile("s_wait_asynccnt 0" ::: "memory");
}

// ---------------------------------------------------------------------------
// Weight conversion: fp32 [K][N] -> bf16 transposed [N][K]
// ---------------------------------------------------------------------------
__global__ void cvt_transpose_kernel(const float* __restrict__ src,
                                     unsigned short* __restrict__ dst,
                                     int K, int N) {
    int idx = blockIdx.x * blockDim.x + threadIdx.x;   // over N*K, dst-major
    if (idx >= K * N) return;
    int k = idx % K, n = idx / K;
    dst[idx] = f2bf(src[(size_t)k * N + n]);
}

__global__ void cvt_copy_kernel(const float* __restrict__ src,
                                unsigned short* __restrict__ dst, int count) {
    int idx = blockIdx.x * blockDim.x + threadIdx.x;
    if (idx < count) dst[idx] = f2bf(src[idx]);
}

// Zero the two pad rows of both padded activation buffers.
__global__ void init_pads_kernel(unsigned short* __restrict__ p1,
                                 unsigned short* __restrict__ p2) {
    int idx = blockIdx.x * blockDim.x + threadIdx.x;   // B*2*E = 8192
    if (idx >= Bn * 2 * En) return;
    int e = idx & (En - 1);
    int r = (idx >> 9) & 1;
    int b = idx >> 10;
    size_t off = ((size_t)b * LP + (r ? (LP - 1) : 0)) * En + e;
    p1[off] = 0; p2[off] = 0;
}

// ---------------------------------------------------------------------------
// Embedding: h = Wtok[x] + Wpos, stored bf16 into padded buffer row l+1.
// ---------------------------------------------------------------------------
__global__ void embed_kernel(const int* __restrict__ x,
                             const float* __restrict__ Wtok,
                             const float* __restrict__ Wpos,
                             unsigned short* __restrict__ Hpad) {
    int idx = blockIdx.x * blockDim.x + threadIdx.x;   // B*L*E = 2^23
    if (idx >= Bn * Ln * En) return;
    int e = idx & (En - 1);
    int l = (idx >> 9) & (Ln - 1);
    int b = idx >> 20;
    int tok = x[b * Ln + l];
    float v = Wtok[(size_t)tok * En + e] + Wpos[(size_t)l * En + e];
    Hpad[((size_t)b * LP + l + 1) * En + e] = f2bf(v);
}

// ---------------------------------------------------------------------------
// Conv1d as GEMM: Y[m,n] = ReLU( sum_K Hpad[row(m,K)] * WcT[n][K] + bias[n] )
// M tile 128 (8 waves x 16), N tile 64 (4 frags/wave), K step 32 (bf16 WMMA).
// A staged via async global->LDS (double buffered); B per-fragment from L2.
// ---------------------------------------------------------------------------
__global__ void conv_gemm_kernel(const unsigned short* __restrict__ Hpad,
                                 const unsigned short* __restrict__ WcT,
                                 const float* __restrict__ bias,
                                 unsigned short* __restrict__ padOut, // may be null
                                 float* __restrict__ fOut) {          // may be null
    __shared__ __align__(16) unsigned short As[2][128 * 32];
    const int tid  = threadIdx.x;
    const int wave = tid >> 5;
    const int lane = tid & 31;
    const int half = lane >> 4;
    const int ln   = lane & 15;
    const int m0blk = blockIdx.x * 128;
    const int n0    = blockIdx.y * 64;
    const int b     = m0blk >> 11;        // 128 | 2048 -> tile within one batch
    const int l0    = m0blk & (Ln - 1);

    const unsigned ldsA[2] = { (unsigned)(size_t)&As[0][0],
                               (unsigned)(size_t)&As[1][0] };

    // two 16B chunks per lane per K-step
    const int chunk0 = wave * 64 + lane;     // + 0 / + 32
    v8f acc[4];
#pragma unroll
    for (int j = 0; j < 4; ++j)
#pragma unroll
        for (int r = 0; r < 8; ++r) acc[j][r] = 0.f;

    // prologue: stage tile s=0 into buffer 0
    {
        const int kconv = 0, c0 = 0;
#pragma unroll
        for (int i = 0; i < 2; ++i) {
            int chunk = chunk0 + i * 32;
            int row = chunk >> 2, seg = chunk & 3;
            const unsigned short* g =
                Hpad + ((size_t)(b * LP) + (l0 + row + kconv)) * En + c0 + seg * 8;
            async_copy16(ldsA[0] + (unsigned)(row * 64 + seg * 16), g);
        }
    }

    for (int s = 0; s < Kc / 32; ++s) {
        wait_async0();        // this wave's staging for tile s complete
        __syncthreads();      // all waves staged tile s; all done reading s-1

        if (s + 1 < Kc / 32) {
            const int kn = (s + 1) >> 4;
            const int cn = ((s + 1) & 15) * 32;
            const unsigned dst = ldsA[(s + 1) & 1];
#pragma unroll
            for (int i = 0; i < 2; ++i) {
                int chunk = chunk0 + i * 32;
                int row = chunk >> 2, seg = chunk & 3;
                const unsigned short* g =
                    Hpad + ((size_t)(b * LP) + (l0 + row + kn)) * En + cn + seg * 8;
                async_copy16(dst + (unsigned)(row * 64 + seg * 16), g);
            }
        }

        // A fragment (16x32 bf16): lanes 0-15 K{0-7,16-23}, lanes 16-31 K{8-15,24-31}
        const unsigned short* Abuf = As[s & 1];
        FragBF a;
        const int abase = (wave * 16 + ln) * 32;
        a.q[0] = *(const uint4*)&Abuf[abase + 8 * half];
        a.q[1] = *(const uint4*)&Abuf[abase + 16 + 8 * half];

#pragma unroll
        for (int j = 0; j < 4; ++j) {
            // B fragment (32x16 bf16): lane n, lanes 0-15 K{0-7,16-23}, lanes 16-31 K{8-15,24-31}
            FragBF bq;
            const unsigned short* wrow =
                WcT + (size_t)(n0 + j * 16 + ln) * Kc + s * 32;
            bq.q[0] = *(const uint4*)(wrow + 8 * half);
            bq.q[1] = *(const uint4*)(wrow + 16 + 8 * half);
            acc[j] = __builtin_amdgcn_wmma_f32_16x16x32_bf16(
                false, a.v, false, bq.v, (short)0, acc[j], false, false);
        }
    }

    // epilogue: bias + ReLU, scatter per C-layout (M = r + 8*half, N = ln)
#pragma unroll
    for (int j = 0; j < 4; ++j) {
        const int col = n0 + j * 16 + ln;
        const float bv = bias[col];
#pragma unroll
        for (int r = 0; r < 8; ++r) {
            int gm = m0blk + wave * 16 + r + 8 * half;
            int l  = gm & (Ln - 1);
            int bb = gm >> 11;
            float v = fmaxf(acc[j][r] + bv, 0.f);
            if (padOut) padOut[((size_t)bb * LP + l + 1) * En + col] = f2bf(v);
            if (fOut)   fOut[(size_t)gm * En + col] = v;
        }
    }
}

// ---------------------------------------------------------------------------
// Forward FFT: 2048-pt radix-2 DIT per (b, e); 4 channels per block in LDS.
// Emits X -> Xw[b][e][f] (float2) and per-(b,e) max |X|^2.
// ---------------------------------------------------------------------------
#define FFT_CH 4

__global__ void fft_fwd_kernel(const float* __restrict__ Hc,
                               float2* __restrict__ Xw,
                               float* __restrict__ maxsq) {
    __shared__ float2 sd[FFT_CH * Ln];    // 64 KB
    const int tid = threadIdx.x;
    const int b   = blockIdx.x;
    const int e0  = blockIdx.y * FFT_CH;

    // load (real input), bit-reversed placement
    for (int idx = tid; idx < Ln * FFT_CH; idx += 256) {
        int l = idx >> 2, ch = idx & 3;
        float v = Hc[((size_t)b * Ln + l) * En + e0 + ch];
        int rv = __brev((unsigned)l) >> 21;           // 11-bit reverse
        sd[ch * Ln + rv] = make_float2(v, 0.f);
    }
    for (int s = 1; s <= 11; ++s) {
        const int hm = 1 << (s - 1);
        __syncthreads();
        for (int t = tid; t < (Ln / 2) * FFT_CH; t += 256) {
            int ch  = t >> 10;
            int j   = t & 1023;
            int grp = j >> (s - 1);
            int pos = j & (hm - 1);
            int i0  = ch * Ln + (grp << s) + pos;
            int i1  = i0 + hm;
            float ang = -3.14159265358979f * (float)pos / (float)hm;
            float c = cosf(ang), sn = sinf(ang);
            float2 u = sd[i0], w = sd[i1];
            float2 tw = make_float2(w.x * c - w.y * sn, w.x * sn + w.y * c);
            sd[i0] = make_float2(u.x + tw.x, u.y + tw.y);
            sd[i1] = make_float2(u.x - tw.x, u.y - tw.y);
        }
    }
    __syncthreads();
    float pm[FFT_CH] = {0.f, 0.f, 0.f, 0.f};
#pragma unroll
    for (int ch = 0; ch < FFT_CH; ++ch)
        for (int f = tid; f < Ln; f += 256) {
            float2 v = sd[ch * Ln + f];
            Xw[((size_t)(b * En + e0 + ch)) * Ln + f] = v;
            pm[ch] = fmaxf(pm[ch], v.x * v.x + v.y * v.y);
        }
    __syncthreads();
    float* scr = (float*)sd;              // reuse LDS for reduction
#pragma unroll
    for (int ch = 0; ch < FFT_CH; ++ch) scr[ch * 256 + tid] = pm[ch];
    __syncthreads();
    for (int off = 128; off > 0; off >>= 1) {
        if (tid < off)
#pragma unroll
            for (int ch = 0; ch < FFT_CH; ++ch)
                scr[ch * 256 + tid] = fmaxf(scr[ch * 256 + tid], scr[ch * 256 + tid + off]);
        __syncthreads();
    }
    if (tid == 0)
#pragma unroll
        for (int ch = 0; ch < FFT_CH; ++ch)
            maxsq[b * En + e0 + ch] = scr[ch * 256];
}

// ---------------------------------------------------------------------------
// Threshold reduction: complexity[b] -> kept-frequency count N[b]; loss = 0.
// ---------------------------------------------------------------------------
__global__ void spectral_n_kernel(const float2* __restrict__ Xw,
                                  const float* __restrict__ maxsq,
                                  int* __restrict__ Nkeep,
                                  float* __restrict__ lossOut) {
    __shared__ int scnt[256];
    const int tid = threadIdx.x;
    const int b   = blockIdx.x;
    int cnt = 0;
    for (int idx = tid; idx < En * Ln; idx += 256) {   // 2^20 per batch
        int e = idx >> 11;
        float2 v = Xw[((size_t)b << 20) + idx];
        float thr2 = 0.01f * maxsq[b * En + e];        // (0.1*max)^2
        cnt += (v.x * v.x + v.y * v.y > thr2) ? 1 : 0;
    }
    scnt[tid] = cnt;
    __syncthreads();
    for (int off = 128; off > 0; off >>= 1) {
        if (tid < off) scnt[tid] += scnt[tid + off];
        __syncthreads();
    }
    if (tid == 0) {
        float complexity = (float)scnt[0] / (float)(En * Ln);
        float ratio = 0.5f * (1.f - complexity);
        ratio = fminf(fmaxf(ratio, 0.1f), 1.f);
        int N = (int)(ratio * (float)Ln);
        N = min(max(N, 1), Ln);
        Nkeep[b] = N;
        if (b == 0) *lossOut = 0.f;  // recon_target == out -> loss is exactly 0
    }
}

// ---------------------------------------------------------------------------
// Masked inverse FFT: keep f < N[b], ifft, take real/L, emit bf16 for GEMM.
// ---------------------------------------------------------------------------
__global__ void fft_inv_kernel(const float2* __restrict__ Xw,
                               const int* __restrict__ Nkeep,
                               unsigned short* __restrict__ XtBf) {
    __shared__ float2 sd[FFT_CH * Ln];
    const int tid = threadIdx.x;
    const int b   = blockIdx.x;
    const int e0  = blockIdx.y * FFT_CH;
    const int N   = Nkeep[b];

    for (int f = tid; f < Ln; f += 256) {
        int rv = __brev((unsigned)f) >> 21;
#pragma unroll
        for (int ch = 0; ch < FFT_CH; ++ch) {
            float2 v = make_float2(0.f, 0.f);
            if (f < N) v = Xw[((size_t)(b * En + e0 + ch)) * Ln + f];
            sd[ch * Ln + rv] = v;
        }
    }
    for (int s = 1; s <= 11; ++s) {
        const int hm = 1 << (s - 1);
        __syncthreads();
        for (int t = tid; t < (Ln / 2) * FFT_CH; t += 256) {
            int ch  = t >> 10;
            int j   = t & 1023;
            int grp = j >> (s - 1);
            int pos = j & (hm - 1);
            int i0  = ch * Ln + (grp << s) + pos;
            int i1  = i0 + hm;
            float ang = 3.14159265358979f * (float)pos / (float)hm;  // inverse
            float c = cosf(ang), sn = sinf(ang);
            float2 u = sd[i0], w = sd[i1];
            float2 tw = make_float2(w.x * c - w.y * sn, w.x * sn + w.y * c);
            sd[i0] = make_float2(u.x + tw.x, u.y + tw.y);
            sd[i1] = make_float2(u.x - tw.x, u.y - tw.y);
        }
    }
    __syncthreads();
    const float inv = 1.f / (float)Ln;
    for (int l = tid; l < Ln; l += 256)
#pragma unroll
        for (int ch = 0; ch < FFT_CH; ++ch)
            XtBf[((size_t)b * Ln + l) * En + e0 + ch] = f2bf(sd[ch * Ln + l].x * inv);
}

// ---------------------------------------------------------------------------
// Projection GEMM: out[m,n] = x_ifft[m,:] . Wp[n,:] + bp[n]   (fp32 out)
// ---------------------------------------------------------------------------
__global__ void proj_gemm_kernel(const unsigned short* __restrict__ A,
                                 const unsigned short* __restrict__ WpBf, // [n][k]
                                 const float* __restrict__ bp,
                                 float* __restrict__ out) {
    __shared__ __align__(16) unsigned short As[2][128 * 32];
    const int tid  = threadIdx.x;
    const int wave = tid >> 5;
    const int lane = tid & 31;
    const int half = lane >> 4;
    const int ln   = lane & 15;
    const int m0blk = blockIdx.x * 128;
    const int n0    = blockIdx.y * 64;

    const unsigned ldsA[2] = { (unsigned)(size_t)&As[0][0],
                               (unsigned)(size_t)&As[1][0] };
    const int chunk0 = wave * 64 + lane;

    v8f acc[4];
#pragma unroll
    for (int j = 0; j < 4; ++j)
#pragma unroll
        for (int r = 0; r < 8; ++r) acc[j][r] = 0.f;

    // prologue: stage tile s=0 into buffer 0
#pragma unroll
    for (int i = 0; i < 2; ++i) {
        int chunk = chunk0 + i * 32;
        int row = chunk >> 2, seg = chunk & 3;
        const unsigned short* g = A + (size_t)(m0blk + row) * En + seg * 8;
        async_copy16(ldsA[0] + (unsigned)(row * 64 + seg * 16), g);
    }

    for (int s = 0; s < En / 32; ++s) {
        wait_async0();
        __syncthreads();

        if (s + 1 < En / 32) {
            const unsigned dst = ldsA[(s + 1) & 1];
#pragma unroll
            for (int i = 0; i < 2; ++i) {
                int chunk = chunk0 + i * 32;
                int row = chunk >> 2, seg = chunk & 3;
                const unsigned short* g =
                    A + (size_t)(m0blk + row) * En + (s + 1) * 32 + seg * 8;
                async_copy16(dst + (unsigned)(row * 64 + seg * 16), g);
            }
        }

        const unsigned short* Abuf = As[s & 1];
        FragBF a;
        const int abase = (wave * 16 + ln) * 32;
        a.q[0] = *(const uint4*)&Abuf[abase + 8 * half];
        a.q[1] = *(const uint4*)&Abuf[abase + 16 + 8 * half];

#pragma unroll
        for (int j = 0; j < 4; ++j) {
            FragBF bq;
            const unsigned short* wrow =
                WpBf + (size_t)(n0 + j * 16 + ln) * En + s * 32;
            bq.q[0] = *(const uint4*)(wrow + 8 * half);
            bq.q[1] = *(const uint4*)(wrow + 16 + 8 * half);
            acc[j] = __builtin_amdgcn_wmma_f32_16x16x32_bf16(
                false, a.v, false, bq.v, (short)0, acc[j], false, false);
        }
    }
#pragma unroll
    for (int j = 0; j < 4; ++j) {
        const int col = n0 + j * 16 + ln;
        const float bv = bp[col];
#pragma unroll
        for (int r = 0; r < 8; ++r) {
            int gm = m0blk + wave * 16 + r + 8 * half;
            out[(size_t)gm * En + col] = acc[j][r] + bv;
        }
    }
}

// ---------------------------------------------------------------------------
// Host-side launch
// ---------------------------------------------------------------------------
static inline size_t align256(size_t v) { return (v + 255) & ~(size_t)255; }

extern "C" void kernel_launch(void* const* d_in, const int* in_sizes, int n_in,
                              void* d_out, int out_size, void* d_ws, size_t ws_size,
                              hipStream_t stream) {
    (void)in_sizes; (void)n_in; (void)out_size; (void)ws_size;
    const int*   x    = (const int*)  d_in[0];
    const float* Wtok = (const float*)d_in[1];
    const float* Wpos = (const float*)d_in[2];
    const float* Wc1  = (const float*)d_in[3];
    const float* bc1  = (const float*)d_in[4];
    const float* Wc2  = (const float*)d_in[5];
    const float* bc2  = (const float*)d_in[6];
    const float* Wp   = (const float*)d_in[7];
    const float* bp   = (const float*)d_in[8];
    float* outp = (float*)d_out;

    char* w = (char*)d_ws;
    size_t off = 0;
    unsigned short* Hpad1 = (unsigned short*)(w + off); off = align256(off + (size_t)Bn * LP * En * 2);
    unsigned short* Hpad2 = (unsigned short*)(w + off); off = align256(off + (size_t)Bn * LP * En * 2);
    float*          Hc    = (float*)         (w + off); off = align256(off + (size_t)Bn * Ln * En * 4);
    float2*         Xw    = (float2*)        (w + off); off = align256(off + (size_t)Bn * En * Ln * 8);
    unsigned short* XtBf  = (unsigned short*)(w + off); off = align256(off + (size_t)Bn * Ln * En * 2);
    unsigned short* WcT1  = (unsigned short*)(w + off); off = align256(off + (size_t)Kc * En * 2);
    unsigned short* WcT2  = (unsigned short*)(w + off); off = align256(off + (size_t)Kc * En * 2);
    unsigned short* WpBf  = (unsigned short*)(w + off); off = align256(off + (size_t)En * En * 2);
    float*          maxsq = (float*)         (w + off); off = align256(off + (size_t)Bn * En * 4);
    int*            Nkeep = (int*)           (w + off); off = align256(off + (size_t)Bn * 4);

    // weights -> bf16 (transposed [N][K] for conv taps; Wp already [out][in])
    cvt_transpose_kernel<<<(Kc * En + 255) / 256, 256, 0, stream>>>(Wc1, WcT1, Kc, En);
    cvt_transpose_kernel<<<(Kc * En + 255) / 256, 256, 0, stream>>>(Wc2, WcT2, Kc, En);
    cvt_copy_kernel<<<(En * En + 255) / 256, 256, 0, stream>>>(Wp, WpBf, En * En);
    init_pads_kernel<<<(Bn * 2 * En + 255) / 256, 256, 0, stream>>>(Hpad1, Hpad2);

    // embedding
    embed_kernel<<<(Bn * Ln * En) / 256, 256, 0, stream>>>(x, Wtok, Wpos, Hpad1);

    // conv1: bf16 padded out; conv2: fp32 out for FFT
    dim3 ggrid(Bn * Ln / 128, En / 64);
    conv_gemm_kernel<<<ggrid, 256, 0, stream>>>(Hpad1, WcT1, bc1, Hpad2, nullptr);
    conv_gemm_kernel<<<ggrid, 256, 0, stream>>>(Hpad2, WcT2, bc2, nullptr, Hc);

    // spectral pipeline
    dim3 fgrid(Bn, En / FFT_CH);
    fft_fwd_kernel<<<fgrid, 256, 0, stream>>>(Hc, Xw, maxsq);
    spectral_n_kernel<<<Bn, 256, 0, stream>>>(Xw, maxsq, Nkeep,
                                              outp + (size_t)Bn * Ln * En);
    fft_inv_kernel<<<fgrid, 256, 0, stream>>>(Xw, Nkeep, XtBf);

    // projection
    proj_gemm_kernel<<<ggrid, 256, 0, stream>>>(XtBf, WpBf, bp, outp);
}
